// AllegroLayer_34531537060549
// MI455X (gfx1250) — matrix operand order repaired and checked by reading, not compiled, not run on hardware
//
#include <hip/hip_runtime.h>
#include <hip/hip_bf16.h>

typedef __attribute__((ext_vector_type(16))) _Float16 v16h;
typedef __attribute__((ext_vector_type(8)))  float    v8f;

#define TILE 16

// ---- path metadata: PATHS = [(0,0,0),(1,1,0),(2,2,0),(0,1,1),(1,0,1),(1,2,1),
//                              (2,1,1),(0,2,2),(1,1,2),(2,0,2),(2,2,2)]
__constant__ int P_L1[11]   = {0,1,2, 0,1,1,2, 0,1,2,2};
__constant__ int P_L2[11]   = {0,1,2, 1,0,2,1, 2,1,0,2};
__constant__ int P_L3[11]   = {0,0,0, 1,1,1,1, 2,2,2,2};
__constant__ int P_SLOT[11] = {0,1,2, 0,1,2,3, 0,1,2,3};

// sparse real-CG tables (COO, packed (i<<8)|(j<<4)|k); Frobenius-normalized
__constant__ int CG_OFF[12] = {0,1,4,9,12,15,24,33,38,47,52,62};
__constant__ unsigned short CG_IJK[62] = {
  0x000,
  0x000,0x110,0x220,
  0x000,0x110,0x220,0x330,0x440,
  0x000,0x011,0x022,
  0x000,0x101,0x202,
  0x000,0x011,0x022,0x110,0x121,0x132,0x220,0x231,0x242,
  0x000,0x101,0x202,0x110,0x211,0x312,0x220,0x321,0x422,
  0x000,0x011,0x022,0x033,0x044,
  0x000,0x011,0x022,0x101,0x112,0x123,0x202,0x213,0x224,
  0x000,0x101,0x202,0x303,0x404,
  0x000,0x111,0x222,0x333,0x444,0x012,0x123,0x234,0x340,0x401
};
__constant__ float CG_V[62] = {
  1.0f,
  0.57735027f,0.57735027f,0.57735027f,
  0.44721360f,0.44721360f,0.44721360f,0.44721360f,0.44721360f,
  0.57735027f,0.57735027f,0.57735027f,
  0.57735027f,0.57735027f,0.57735027f,
  0.33333334f,0.33333334f,0.33333334f,0.33333334f,0.33333334f,0.33333334f,0.33333334f,0.33333334f,0.33333334f,
  0.33333334f,0.33333334f,0.33333334f,0.33333334f,0.33333334f,0.33333334f,0.33333334f,0.33333334f,0.33333334f,
  0.44721360f,0.44721360f,0.44721360f,0.44721360f,0.44721360f,
  0.33333334f,0.33333334f,0.33333334f,0.33333334f,0.33333334f,0.33333334f,0.33333334f,0.33333334f,0.33333334f,
  0.44721360f,0.44721360f,0.44721360f,0.44721360f,0.44721360f,
  0.31622776f,0.31622776f,0.31622776f,0.31622776f,0.31622776f,
  0.31622776f,0.31622776f,0.31622776f,0.31622776f,0.31622776f
};

__constant__ int OFFL_c[3] = {0, 16, 64};      // l-block offsets inside 144
__constant__ int XW_c[3]   = {48, 64, 64};     // per-l3 concatenated widths
__constant__ int XB_c[3]   = {0, 768, 3840};   // per-l3 base in X buffer (floats)

// WMMA f16 A-operand swizzle (wave32): element (M,K) -> half index in A-staging
// buffer laid out as [kb][lane][16 halves], contiguous k order: K = kb*32 + 16*(lane>>4) + h.
__device__ __forceinline__ int a_swz(int M, int K) {
  int kb = K >> 5, r = K & 31;
  int lane = M + ((r >> 4) << 4);
  return (kb * 32 + lane) * 16 + (r & 15);
}

// ---------------- helper kernels ----------------
__global__ void zero_kernel(float* p, int n) {
  int i = blockIdx.x * 256 + threadIdx.x;
  if (i < n) p[i] = 0.f;
}

__global__ void scatter_kernel(const float* __restrict__ ang,
                               const int* __restrict__ eidx,
                               float* __restrict__ nodeY, int E) {
  int i = blockIdx.x * 256 + threadIdx.x;
  if (i < E * 9) {
    int e = i / 9, j = i - e * 9;
    int s = eidx[e * 2 + 0];
    atomicAdd(&nodeY[s * 9 + j], ang[e * 9 + j]);
  }
}

// Convert MLP weights to f16, pre-swizzled into a contiguous-k wave32 WMMA B-operand
// layout: buffer[((kb*NT + nt)*32 + lane)*16 + h] holds W[K][N] with
// K = kb*32 + 16*(lane>>4) + h,  N = nt*16 + (lane&15).
// W1: K padded 144->160 (5 k-steps, 16 n-tiles); W2: 256x128 (8 k-steps, 8 n-tiles)
__global__ void prep_weights(const float* __restrict__ W1, const float* __restrict__ W2,
                             _Float16* __restrict__ W1s, _Float16* __restrict__ W2s) {
  int i = blockIdx.x * 256 + threadIdx.x;
  if (i < 160 * 256) {
    int h = i & 15, lane = (i >> 4) & 31, nt = (i >> 9) & 15, kb = i >> 13;
    int K = kb * 32 + ((lane >> 4) << 4) + h;
    int N = nt * 16 + (lane & 15);
    W1s[i] = (_Float16)((K < 144) ? W1[K * 256 + N] : 0.f);
  } else {
    int j = i - 160 * 256;
    if (j < 256 * 128) {
      int h = j & 15, lane = (j >> 4) & 31, nt = (j >> 9) & 7, kb = j >> 12;
      int K = kb * 32 + ((lane >> 4) << 4) + h;
      int N = nt * 16 + (lane & 15);
      W2s[j] = (_Float16)W2[K * 128 + N];
    }
  }
}

// ---------------- fused main kernel: one 16-edge tile per block ----------------
__global__ __launch_bounds__(256)
void allegro_main(const float* __restrict__ scal, const float* __restrict__ tens,
                  const float* __restrict__ envW, const float* __restrict__ envb,
                  const float* __restrict__ tpw,
                  const float* __restrict__ lw0, const float* __restrict__ lw1,
                  const float* __restrict__ lw2,
                  const float* __restrict__ b1v, const float* __restrict__ b2v,
                  const float* __restrict__ tenvW, const float* __restrict__ tenvb,
                  const int* __restrict__ eidx, const float* __restrict__ nodeY,
                  const _Float16* __restrict__ W1s, const _Float16* __restrict__ W2s,
                  float* __restrict__ out, int E) {
  // 59.6 KB LDS pool with lifetime-overlapped regions
  __shared__ __align__(32) char pool[60992];
  float*    Xbuf = (float*)(pool + 0);          // 8960 f: TP outputs  (phase 1-2)
  float*    upd  = (float*)(pool + 0);          // 16x128 f: updated scalars (phase 3+)
  float*    tS   = (float*)(pool + 35840);      // 16x144 f: tensor tile (phase 1-2)
  _Float16* hA   = (_Float16*)(pool + 35840);   // 8kb x 32 x 16 h: hidden, A-swizzled
  _Float16* comb = (_Float16*)(pool + 45056);   // 5kb x 32 x 16 h: combined, A-swizzled
  float*    nT   = (float*)(pool + 50176);      // 16x144 f: new tensor blocks
  float*    envB = (float*)(pool + 59392);      // 16x16 f: env_w, later env_out
  float*    eYs  = (float*)(pool + 60416);      // 16x9 f: gathered angular

  const int tid  = threadIdx.x;
  const int lane = tid & 31;
  const int wave = tid >> 5;
  const int e0   = blockIdx.x * TILE;

  // ---- phase 0: loads, env_w, gather eY, scalar->f16 (A-swizzled)
  for (int o = tid; o < TILE * 128; o += 256) {
    int m = o >> 7, k = o & 127;
    int ge = e0 + m; if (ge >= E) ge = E - 1;
    comb[a_swz(m, k)] = (_Float16)scal[ge * 128 + k];
  }
  { // zero the kb=4 block (K=128..159); invariants land there in phase 2
    comb[(4 * 32) * 16 + tid] = (_Float16)0.f;
    comb[(4 * 32) * 16 + 256 + tid] = (_Float16)0.f;
  }
  for (int o = tid; o < TILE * 144; o += 256) {
    int m = o / 144, k = o - m * 144;
    int ge = e0 + m; if (ge >= E) ge = E - 1;
    tS[o] = tens[ge * 144 + k];
  }
  if (tid < TILE * 9) {
    int m = tid / 9, j = tid - m * 9;
    int ge = e0 + m; if (ge >= E) ge = E - 1;
    int s = eidx[ge * 2 + 0];
    eYs[m * 9 + j] = nodeY[s * 9 + j] * 0.17677669529663687f;  // /sqrt(32)
  }
  {
    int m = tid >> 4, c = tid & 15;
    int ge = e0 + m; if (ge >= E) ge = E - 1;
    float acc = envb[c];
    for (int k = 0; k < 128; ++k) acc += scal[ge * 128 + k] * envW[k * 16 + c];
    envB[tid] = acc;
  }
  __syncthreads();

  // ---- phase 1: CG tensor product, scaled by env_w and tp_w -> Xbuf
  for (int p = 0; p < 11; ++p) {
    int l1 = P_L1[p], l2 = P_L2[p], l3 = P_L3[p], slot = P_SLOT[p];
    int deg3 = 2 * l3 + 1;
    int W = XW_c[l3];
    int nout = TILE * deg3 * 16;
    for (int o = tid; o < nout; o += 256) {
      int m = o / (deg3 * 16);
      int r = o - m * deg3 * 16;
      int k = r >> 4, c = r & 15;
      float acc = 0.f;
      for (int t = CG_OFF[p]; t < CG_OFF[p + 1]; ++t) {
        unsigned ee = CG_IJK[t];
        if ((int)(ee & 15u) != k) continue;
        int i = (ee >> 8) & 15, j = (ee >> 4) & 15;
        acc += tS[m * 144 + OFFL_c[l1] + i * 16 + c] * eYs[m * 9 + l2 * l2 + j] * CG_V[t];
      }
      acc *= envB[m * 16 + c] * tpw[p * 16 + c];
      Xbuf[XB_c[l3] + (m * deg3 + k) * W + slot * 16 + c] = acc;
    }
  }
  __syncthreads();

  // ---- phase 2: per-l linear mixing -> nT, invariants -> comb (K=128..143)
  for (int o = tid; o < TILE * 144; o += 256) {
    int m = o / 144, idx = o - m * 144;
    int l = (idx < 16) ? 0 : ((idx < 64) ? 1 : 2);
    int base = OFFL_c[l];
    int k = (idx - base) >> 4, c = (idx - base) & 15;
    int W = XW_c[l];
    const float* lw = (l == 0) ? lw0 : ((l == 1) ? lw1 : lw2);
    const float* xr = &Xbuf[XB_c[l] + (m * (2 * l + 1) + k) * W];
    float acc = 0.f;
    for (int p = 0; p < W; ++p) acc += xr[p] * lw[p * 16 + c];
    acc *= rsqrtf((float)W);
    nT[o] = acc;
    if (idx < 16) comb[a_swz(m, 128 + idx)] = (_Float16)acc;
  }
  __syncthreads();

  // ---- phase 3: MLP1 via WMMA: (16x160 f16) @ (160x256 f16), silu -> hA
  {
    const v16h* combV = (const v16h*)comb;
    const v16h* W1v   = (const v16h*)W1s;
    for (int nt = wave; nt < 16; nt += 8) {
      v8f acc = {0.f, 0.f, 0.f, 0.f, 0.f, 0.f, 0.f, 0.f};
      int Ncol = nt * 16 + (lane & 15);
      for (int kb = 0; kb < 5; ++kb) {
        v16h a = combV[kb * 32 + lane];
        v16h b = W1v[(kb * 16 + nt) * 32 + lane];
        acc = __builtin_amdgcn_wmma_f32_16x16x32_f16(false, a, false, b, (short)0, acc,
                                                     false, false);
      }
      float bias = b1v[Ncol];
#pragma unroll
      for (int r = 0; r < 8; ++r) {
        int Mo = r + ((lane >> 4) ? 8 : 0);
        float val = acc[r] + bias;
        float sv = val * __builtin_amdgcn_rcpf(1.f + __expf(-val));  // fast silu
        hA[a_swz(Mo, Ncol)] = (_Float16)sv;
      }
    }
  }
  __syncthreads();

  // ---- phase 4: MLP2 via WMMA: (16x256 f16) @ (256x128 f16) + residual
  {
    const v16h* hV  = (const v16h*)hA;
    const v16h* W2v = (const v16h*)W2s;
    int nt = wave;  // 8 waves cover 128 cols
    int col = nt * 16 + (lane & 15);
    v8f acc = {0.f, 0.f, 0.f, 0.f, 0.f, 0.f, 0.f, 0.f};
    for (int kb = 0; kb < 8; ++kb) {
      v16h a = hV[kb * 32 + lane];
      v16h b = W2v[(kb * 8 + nt) * 32 + lane];
      acc = __builtin_amdgcn_wmma_f32_16x16x32_f16(false, a, false, b, (short)0, acc,
                                                   false, false);
    }
    float bias = b2v[col];
#pragma unroll
    for (int r = 0; r < 8; ++r) {
      int Mo = r + ((lane >> 4) ? 8 : 0);
      int ge = e0 + Mo;
      float val = acc[r] + bias;
      float sc = (ge < E) ? scal[ge * 128 + col] : 0.f;
      float u = 0.89442719f * sc + 0.44721360f * val;  // RES_A, RES_B (alpha=0.5)
      upd[Mo * 128 + col] = u;
      if (ge < E) out[(size_t)ge * 128 + col] = u;
    }
  }
  __syncthreads();

  // ---- phase 5: env_out = updated_scalars @ tenv_W + tenv_b
  {
    int m = tid >> 4, c = tid & 15;
    float acc = tenvb[c];
    for (int k = 0; k < 128; ++k) acc += upd[m * 128 + k] * tenvW[k * 16 + c];
    envB[tid] = acc;
  }
  __syncthreads();

  // ---- phase 6: updated_tensor = new_blocks * env_out (channel-wise)
  {
    float* outT = out + (size_t)E * 128;
    for (int o = tid; o < TILE * 144; o += 256) {
      int m = o / 144, idx = o - m * 144;
      int ge = e0 + m;
      if (ge < E) outT[(size_t)ge * 144 + idx] = nT[o] * envB[m * 16 + (idx & 15)];
    }
  }
}

// ---------------- launcher ----------------
extern "C" void kernel_launch(void* const* d_in, const int* in_sizes, int n_in,
                              void* d_out, int out_size, void* d_ws, size_t ws_size,
                              hipStream_t stream) {
  const float* scal  = (const float*)d_in[0];
  const float* tens  = (const float*)d_in[1];
  const float* ang   = (const float*)d_in[2];
  const float* envW  = (const float*)d_in[3];
  const float* envb  = (const float*)d_in[4];
  const float* tpw   = (const float*)d_in[5];
  const float* lw0   = (const float*)d_in[6];
  const float* lw1   = (const float*)d_in[7];
  const float* lw2   = (const float*)d_in[8];
  const float* W1    = (const float*)d_in[9];
  const float* b1v   = (const float*)d_in[10];
  const float* W2    = (const float*)d_in[11];
  const float* b2v   = (const float*)d_in[12];
  const float* tenvW = (const float*)d_in[13];
  const float* tenvb = (const float*)d_in[14];
  const int*   eidx  = (const int*)d_in[15];

  const int E  = in_sizes[0] / 128;
  const int NY = 20000 * 9;  // n_nodes * SH_DIM (reference constant)

  // workspace: nodeY (720 KB) | W1s swizzled 160x256 f16 | W2s swizzled 256x128 f16
  float*    nodeY = (float*)d_ws;
  _Float16* W1s   = (_Float16*)((char*)d_ws + 720896);
  _Float16* W2s   = (_Float16*)((char*)d_ws + 720896 + 160 * 256 * 2);
  float*    out   = (float*)d_out;

  zero_kernel<<<(NY + 255) / 256, 256, 0, stream>>>(nodeY, NY);
  scatter_kernel<<<(E * 9 + 255) / 256, 256, 0, stream>>>(ang, eidx, nodeY, E);
  prep_weights<<<(160 * 256 + 256 * 128 + 255) / 256, 256, 0, stream>>>(W1, W2, W1s, W2s);
  allegro_main<<<(E + TILE - 1) / TILE, 256, 0, stream>>>(
      scal, tens, envW, envb, tpw, lw0, lw1, lw2, b1v, b2v, tenvW, tenvb,
      eidx, nodeY, W1s, W2s, out, E);
}
